// CausalMaskMultiHeadSelfAttention_12936441495628
// MI455X (gfx1250) — matrix operand order repaired and checked by reading, not compiled, not run on hardware
//
#include <hip/hip_runtime.h>

#define B_ 4
#define S_ 2048
#define D_ 768
#define H_ 12
#define DK_ 64

typedef __bf16 bf16_t;
typedef __attribute__((ext_vector_type(16))) __bf16 v16bf;
typedef __attribute__((ext_vector_type(8)))  __bf16 v8bf;
typedef __attribute__((ext_vector_type(2)))  __bf16 v2bf;
typedef __attribute__((ext_vector_type(8)))  float  v8f;

// Build a v16bf A/B fragment from two contiguous 16-byte chunks.
__device__ __forceinline__ v16bf frag16(const bf16_t* lo, const bf16_t* hi) {
    v8bf a = *(const v8bf*)lo;
    v8bf b = *(const v8bf*)hi;
    v16bf r;
#pragma unroll
    for (int i = 0; i < 8; ++i) { r[i] = a[i]; r[i + 8] = b[i]; }
    return r;
}

__device__ __forceinline__ v8f wmma_bf16(v16bf a, v16bf b, v8f c) {
    return __builtin_amdgcn_wmma_f32_16x16x32_bf16(
        /*neg_a=*/false, a, /*neg_b=*/false, b,
        /*c_mod=*/(short)0, c, /*reuse_a=*/false, /*reuse_b=*/false);
}

// Async 16-byte copy global -> LDS (ASYNCcnt tracked, no VGPR roundtrip).
// LDS address = low 32 bits of the generic pointer (ISA 10.2 aperture rule).
__device__ __forceinline__ void async_copy16(void* lds, const void* gmem) {
    unsigned ldsaddr = (unsigned)(unsigned long long)lds;
    unsigned long long gaddr = (unsigned long long)gmem;
    asm volatile("global_load_async_to_lds_b128 %0, %1, off"
                 :: "v"(ldsaddr), "v"(gaddr)
                 : "memory");
}
__device__ __forceinline__ void wait_async0() {
    asm volatile("s_wait_asynccnt 0" ::: "memory");
}

// ---------------------------------------------------------------------------
// Kernel 1: QKV projection.  C = X @ W_t^T  (t = 0:Q, 1:K, 2:V)
// Block tile 128x128, 8 waves, wave tile 32x64, K step 64 (bf16 WMMA).
// Q scaled by 1/sqrt(dk)=0.125 at write.  V written transposed (b,h,dk,s).
// ---------------------------------------------------------------------------
__global__ __launch_bounds__(256) void qkv_proj_kernel(
    const float* __restrict__ X, const float* __restrict__ Wqkv,
    bf16_t* __restrict__ Q, bf16_t* __restrict__ K, bf16_t* __restrict__ VT)
{
    __shared__ __align__(16) bf16_t As[128 * 64];
    __shared__ __align__(16) bf16_t Bs[128 * 64];

    const int tid  = threadIdx.x;
    const int lane = tid & 31;
    const int wave = tid >> 5;
    const int wm   = wave >> 1;   // 0..3  (M direction)
    const int wn   = wave & 1;    // 0..1  (N direction)
    const int n0   = blockIdx.x * 128;
    const int m0   = blockIdx.y * 128;
    const int t    = blockIdx.z;
    const float* Wt = Wqkv + (size_t)t * D_ * D_;

    const int srow  = tid >> 1;        // 0..127 staging row
    const int shalf = (tid & 1) * 32;  // 0 or 32 (k offset)

    v8f acc[2][4];
#pragma unroll
    for (int i = 0; i < 2; ++i)
#pragma unroll
        for (int j = 0; j < 4; ++j)
#pragma unroll
            for (int r = 0; r < 8; ++r) acc[i][j][r] = 0.f;

    for (int k0 = 0; k0 < D_; k0 += 64) {
        // Stage A tile (X, fp32 -> bf16): 32 values per thread
        {
            const float* g = X + (size_t)(m0 + srow) * D_ + k0 + shalf;
            float tmp[32];
#pragma unroll
            for (int c = 0; c < 8; ++c)
                *(float4*)&tmp[c * 4] = ((const float4*)g)[c];
#pragma unroll
            for (int c = 0; c < 4; ++c) {
                v8bf o;
#pragma unroll
                for (int i = 0; i < 8; ++i) o[i] = (bf16_t)tmp[c * 8 + i];
                *(v8bf*)&As[srow * 64 + shalf + c * 8] = o;
            }
        }
        // Stage B tile (W rows = output features, fp32 -> bf16)
        {
            const float* g = Wt + (size_t)(n0 + srow) * D_ + k0 + shalf;
            float tmp[32];
#pragma unroll
            for (int c = 0; c < 8; ++c)
                *(float4*)&tmp[c * 4] = ((const float4*)g)[c];
#pragma unroll
            for (int c = 0; c < 4; ++c) {
                v8bf o;
#pragma unroll
                for (int i = 0; i < 8; ++i) o[i] = (bf16_t)tmp[c * 8 + i];
                *(v8bf*)&Bs[srow * 64 + shalf + c * 8] = o;
            }
        }
        __syncthreads();

        v16bf af[2][2];   // [m-subtile][k-step]
        {
            const int koff = (lane < 16) ? 0 : 8;
#pragma unroll
            for (int im = 0; im < 2; ++im)
#pragma unroll
                for (int ik = 0; ik < 2; ++ik) {
                    const bf16_t* base =
                        &As[(wm * 32 + im * 16 + (lane & 15)) * 64 + ik * 32];
                    af[im][ik] = frag16(base + koff, base + koff + 16);
                }
        }
#pragma unroll
        for (int j = 0; j < 4; ++j) {
            const int col = wn * 64 + j * 16 + (lane & 15);
#pragma unroll
            for (int ik = 0; ik < 2; ++ik) {
                const int koffB = ik * 32 + ((lane < 16) ? 0 : 8);
                const bf16_t* base = &Bs[col * 64 + koffB];
                v16bf bfrag = frag16(base, base + 16);
#pragma unroll
                for (int im = 0; im < 2; ++im)
                    acc[im][j] = wmma_bf16(af[im][ik], bfrag, acc[im][j]);
            }
        }
        __syncthreads();
    }

    // Epilogue: uniform branch on t hoisted out of the store loops.
    if (t < 2) {
        bf16_t* dst = (t == 0) ? Q : K;
        const float scale = (t == 0) ? 0.125f : 1.0f;
#pragma unroll
        for (int i = 0; i < 2; ++i) {
#pragma unroll
            for (int j = 0; j < 4; ++j) {
                const int col = n0 + wn * 64 + j * 16 + (lane & 15);
                const int h   = col >> 6;
                const int dd  = col & 63;
                const int row0 = m0 + wm * 32 + i * 16 + ((lane < 16) ? 0 : 8);
                const int bb   = row0 >> 11;
                const int ss0  = row0 & 2047;
                const size_t base =
                    (((size_t)bb * H_ + h) * S_ + ss0) * DK_ + dd;
#pragma unroll
                for (int r = 0; r < 8; ++r)
                    dst[base + (size_t)r * DK_] = (bf16_t)(acc[i][j][r] * scale);
            }
        }
    } else {
        // V^T: adjacent r are adjacent in memory -> packed 2x bf16 stores
#pragma unroll
        for (int i = 0; i < 2; ++i) {
#pragma unroll
            for (int j = 0; j < 4; ++j) {
                const int col = n0 + wn * 64 + j * 16 + (lane & 15);
                const int h   = col >> 6;
                const int dd  = col & 63;
                const int row0 = m0 + wm * 32 + i * 16 + ((lane < 16) ? 0 : 8);
                const int bb   = row0 >> 11;
                const int ss0  = row0 & 2047;
                const size_t base =
                    (((size_t)bb * H_ + h) * DK_ + dd) * S_ + ss0;
#pragma unroll
                for (int r = 0; r < 8; r += 2) {
                    v2bf p;
                    p[0] = (bf16_t)acc[i][j][r];
                    p[1] = (bf16_t)acc[i][j][r + 1];
                    *(v2bf*)&VT[base + r] = p;
                }
            }
        }
    }
}

// ---------------------------------------------------------------------------
// Kernel 2: flash attention per (b, h, 64-row q tile).  128 threads = 4 waves,
// each wave owns 16 query rows.  Online softmax, causal mask on diagonal tile.
// K / V^T tiles staged with GLOBAL_LOAD_ASYNC_TO_LDS_B128.
// ---------------------------------------------------------------------------
__global__ __launch_bounds__(128) void attn_kernel(
    const bf16_t* __restrict__ Q, const bf16_t* __restrict__ K,
    const bf16_t* __restrict__ VT, bf16_t* __restrict__ O)
{
    __shared__ __align__(16) bf16_t Ks[64 * 64];
    __shared__ __align__(16) bf16_t Vs[64 * 64];      // V^T tile: [dv][key]
    __shared__ __align__(16) bf16_t Ps[4][16 * 64];   // wave-private P scratch

    const int tid  = threadIdx.x;
    const int lane = tid & 31;
    const int wave = tid >> 5;
    const int qt = blockIdx.x;          // 0..31
    const int h  = blockIdx.y;
    const int bb = blockIdx.z;
    const size_t headoff = (size_t)bb * H_ + h;
    const bf16_t* Qg = Q  + headoff * S_ * DK_;
    const bf16_t* Kg = K  + headoff * S_ * DK_;
    const bf16_t* Vg = VT + headoff * DK_ * S_;

    // Q fragments for this wave's 16 rows, held in registers for all key tiles.
    v16bf qf[2];
    {
        const int row  = qt * 64 + wave * 16 + (lane & 15);
        const int koff = (lane < 16) ? 0 : 8;
#pragma unroll
        for (int i = 0; i < 2; ++i) {
            const bf16_t* base = Qg + (size_t)row * DK_ + i * 32 + koff;
            qf[i] = frag16(base, base + 16);
        }
    }

    v8f oacc[4];
#pragma unroll
    for (int j = 0; j < 4; ++j)
#pragma unroll
        for (int r = 0; r < 8; ++r) oacc[j][r] = 0.f;
    float mrow[8], lrow[8];
#pragma unroll
    for (int r = 0; r < 8; ++r) { mrow[r] = -1e30f; lrow[r] = 0.f; }

    const int skey  = tid >> 1;         // staging row 0..63
    const int shalf = (tid & 1) * 32;   // 0 or 32

    for (int kt = 0; kt <= qt; ++kt) {
        // Stage K tile [key][d] and V^T tile [dv][key] via async copies
        {
            const bf16_t* gk = Kg + (size_t)(kt * 64 + skey) * DK_ + shalf;
            bf16_t* sk = &Ks[skey * 64 + shalf];
            const bf16_t* gv = Vg + (size_t)skey * S_ + kt * 64 + shalf;
            bf16_t* sv = &Vs[skey * 64 + shalf];
#pragma unroll
            for (int c = 0; c < 4; ++c) async_copy16(sk + c * 8, gk + c * 8);
#pragma unroll
            for (int c = 0; c < 4; ++c) async_copy16(sv + c * 8, gv + c * 8);
        }
        wait_async0();
        __syncthreads();

        // S = Q @ K^T  (Q pre-scaled by 1/sqrt(dk))
        v8f sacc[4];
#pragma unroll
        for (int j = 0; j < 4; ++j)
#pragma unroll
            for (int r = 0; r < 8; ++r) sacc[j][r] = 0.f;
#pragma unroll
        for (int j = 0; j < 4; ++j) {
            const int keyr = j * 16 + (lane & 15);
#pragma unroll
            for (int i = 0; i < 2; ++i) {
                const int koffB = i * 32 + ((lane < 16) ? 0 : 8);
                const bf16_t* base = &Ks[keyr * 64 + koffB];
                v16bf bfrag = frag16(base, base + 16);
                sacc[j] = wmma_bf16(qf[i], bfrag, sacc[j]);
            }
        }

        // Causal mask on diagonal tile
        if (kt == qt) {
#pragma unroll
            for (int j = 0; j < 4; ++j) {
                const int colg = j * 16 + (lane & 15);
#pragma unroll
                for (int r = 0; r < 8; ++r) {
                    const int rowl = wave * 16 + ((lane < 16) ? r : r + 8);
                    if (colg > rowl) sacc[j][r] = -1e30f;
                }
            }
        }

        // Online softmax per row (cross-lane reduce within 16-lane halves)
#pragma unroll
        for (int r = 0; r < 8; ++r) {
            float mx = -1e30f;
#pragma unroll
            for (int j = 0; j < 4; ++j) mx = fmaxf(mx, sacc[j][r]);
#pragma unroll
            for (int off = 8; off >= 1; off >>= 1)
                mx = fmaxf(mx, __shfl_xor(mx, off, 32));
            const float mnew = fmaxf(mrow[r], mx);
            const float corr = __expf(mrow[r] - mnew);
            float rs = 0.f;
#pragma unroll
            for (int j = 0; j < 4; ++j) {
                const float p = __expf(sacc[j][r] - mnew);
                sacc[j][r] = p;
                rs += p;
            }
#pragma unroll
            for (int off = 8; off >= 1; off >>= 1)
                rs += __shfl_xor(rs, off, 32);
            lrow[r] = lrow[r] * corr + rs;
            mrow[r] = mnew;
#pragma unroll
            for (int j = 0; j < 4; ++j) oacc[j][r] *= corr;
        }

        // Re-lay P through wave-private LDS into A-fragment form
#pragma unroll
        for (int j = 0; j < 4; ++j)
#pragma unroll
            for (int r = 0; r < 8; ++r)
                Ps[wave][((lane < 16) ? r : r + 8) * 64 + j * 16 + (lane & 15)] =
                    (bf16_t)sacc[j][r];

        // O += P @ V
#pragma unroll
        for (int i = 0; i < 2; ++i) {
            const int koff = (lane < 16) ? 0 : 8;
            const bf16_t* abase = &Ps[wave][(lane & 15) * 64 + i * 32 + koff];
            v16bf pfrag = frag16(abase, abase + 16);
#pragma unroll
            for (int j = 0; j < 4; ++j) {
                const int dv    = j * 16 + (lane & 15);
                const int koffB = i * 32 + ((lane < 16) ? 0 : 8);
                const bf16_t* bbase = &Vs[dv * 64 + koffB];
                v16bf vfrag = frag16(bbase, bbase + 16);
                oacc[j] = wmma_bf16(pfrag, vfrag, oacc[j]);
            }
        }
        __syncthreads();
    }

    // Normalize and write O in (b, s, h*64+dv) bf16 layout
#pragma unroll
    for (int j = 0; j < 4; ++j) {
        const int dv = j * 16 + (lane & 15);
#pragma unroll
        for (int r = 0; r < 8; ++r) {
            const int rowl = wave * 16 + ((lane < 16) ? r : r + 8);
            const int srow = qt * 64 + rowl;
            const float inv = 1.f / lrow[r];
            O[((size_t)bb * S_ + srow) * D_ + h * DK_ + dv] = (bf16_t)(oacc[j][r] * inv);
        }
    }
}

// ---------------------------------------------------------------------------
// Kernel 3: output projection.  out(fp32) = O_bf16 @ Wo^T.  K step 64.
// A tile staged with async copies (already bf16); B tile converted fp32->bf16.
// ---------------------------------------------------------------------------
__global__ __launch_bounds__(256) void out_proj_kernel(
    const bf16_t* __restrict__ A, const float* __restrict__ Wo,
    float* __restrict__ out)
{
    __shared__ __align__(16) bf16_t As[128 * 64];
    __shared__ __align__(16) bf16_t Bs[128 * 64];

    const int tid  = threadIdx.x;
    const int lane = tid & 31;
    const int wave = tid >> 5;
    const int wm   = wave >> 1;
    const int wn   = wave & 1;
    const int n0   = blockIdx.x * 128;
    const int m0   = blockIdx.y * 128;

    const int srow  = tid >> 1;
    const int shalf = (tid & 1) * 32;

    v8f acc[2][4];
#pragma unroll
    for (int i = 0; i < 2; ++i)
#pragma unroll
        for (int j = 0; j < 4; ++j)
#pragma unroll
            for (int r = 0; r < 8; ++r) acc[i][j][r] = 0.f;

    for (int k0 = 0; k0 < D_; k0 += 64) {
        // Stage A tile (already bf16): async 16B copies
        {
            const bf16_t* g = A + (size_t)(m0 + srow) * D_ + k0 + shalf;
            bf16_t* s = &As[srow * 64 + shalf];
#pragma unroll
            for (int c = 0; c < 4; ++c) async_copy16(s + c * 8, g + c * 8);
        }
        // Stage B tile (Wo, fp32 -> bf16)
        {
            const float* g = Wo + (size_t)(n0 + srow) * D_ + k0 + shalf;
            float tmp[32];
#pragma unroll
            for (int c = 0; c < 8; ++c)
                *(float4*)&tmp[c * 4] = ((const float4*)g)[c];
#pragma unroll
            for (int c = 0; c < 4; ++c) {
                v8bf o;
#pragma unroll
                for (int i = 0; i < 8; ++i) o[i] = (bf16_t)tmp[c * 8 + i];
                *(v8bf*)&Bs[srow * 64 + shalf + c * 8] = o;
            }
        }
        wait_async0();
        __syncthreads();

        v16bf af[2][2];
        {
            const int koff = (lane < 16) ? 0 : 8;
#pragma unroll
            for (int im = 0; im < 2; ++im)
#pragma unroll
                for (int ik = 0; ik < 2; ++ik) {
                    const bf16_t* base =
                        &As[(wm * 32 + im * 16 + (lane & 15)) * 64 + ik * 32];
                    af[im][ik] = frag16(base + koff, base + koff + 16);
                }
        }
#pragma unroll
        for (int j = 0; j < 4; ++j) {
            const int col = wn * 64 + j * 16 + (lane & 15);
#pragma unroll
            for (int ik = 0; ik < 2; ++ik) {
                const int koffB = ik * 32 + ((lane < 16) ? 0 : 8);
                const bf16_t* base = &Bs[col * 64 + koffB];
                v16bf bfrag = frag16(base, base + 16);
#pragma unroll
                for (int im = 0; im < 2; ++im)
                    acc[im][j] = wmma_bf16(af[im][ik], bfrag, acc[im][j]);
            }
        }
        __syncthreads();
    }

#pragma unroll
    for (int i = 0; i < 2; ++i) {
#pragma unroll
        for (int j = 0; j < 4; ++j) {
            const int col = n0 + wn * 64 + j * 16 + (lane & 15);
            const int row0 = m0 + wm * 32 + i * 16 + ((lane < 16) ? 0 : 8);
            float* dst = out + (size_t)row0 * D_ + col;
#pragma unroll
            for (int r = 0; r < 8; ++r)
                dst[(size_t)r * D_] = acc[i][j][r];
        }
    }
}

// ---------------------------------------------------------------------------
extern "C" void kernel_launch(void* const* d_in, const int* in_sizes, int n_in,
                              void* d_out, int out_size, void* d_ws, size_t ws_size,
                              hipStream_t stream) {
    const float* X    = (const float*)d_in[0];   // (4, 2048, 768)
    const float* Wqkv = (const float*)d_in[1];   // (3, 768, 768)
    const float* Wo   = (const float*)d_in[2];   // (768, 768)
    float* out = (float*)d_out;                  // (4, 2048, 768)

    const size_t qkvElems = (size_t)B_ * H_ * S_ * DK_;  // 6,291,456
    bf16_t* Qb  = (bf16_t*)d_ws;
    bf16_t* Kb  = Qb  + qkvElems;
    bf16_t* VTb = Kb  + qkvElems;
    bf16_t* Ob  = VTb + qkvElems;                // total 48 MB of workspace

    // 1) QKV projection (z = t in {Q, K, V})
    qkv_proj_kernel<<<dim3(D_ / 128, (B_ * S_) / 128, 3), 256, 0, stream>>>(
        X, Wqkv, Qb, Kb, VTb);
    // 2) Flash attention per (q-tile, head, batch)
    attn_kernel<<<dim3(S_ / 64, H_, B_), 128, 0, stream>>>(Qb, Kb, VTb, Ob);
    // 3) Output projection
    out_proj_kernel<<<dim3(D_ / 128, (B_ * S_) / 128), 256, 0, stream>>>(
        Ob, Wo, out);
}
